// BasedAttention_82214263980185
// MI455X (gfx1250) — compile-verified, matrix operand order checked by reading, not run on hardware
//
#include <hip/hip_runtime.h>
#include <hip/hip_bf16.h>
#include <math.h>

typedef __attribute__((ext_vector_type(16))) _Float16 v16h;
typedef __attribute__((ext_vector_type(8)))  _Float16 v8h;
typedef __attribute__((ext_vector_type(8)))  float    v8f;

#define D_MODEL 1024
#define N_HEADS 16
#define HEAD_DIM 64
#define FEAT 16
#define WINDOW 64
#define FPAD 160      // 1 + 16 + 136 = 153, padded to 160 (5 K-steps of 32)
#define DAUG 80       // 64 v-dims + ones column (idx 64) + zero pad
#define CHUNK 128
#define TSEQ 2048
#define BATCH 2
#define NROWS (BATCH*TSEQ)   // 4096

// ---- WMMA helpers ----------------------------------------------------------

static __device__ __forceinline__ v8f wmma16(v16h a, v16h b, v8f c) {
  return __builtin_amdgcn_wmma_f32_16x16x32_f16(false, a, false, b, (short)0, c,
                                                false, false);
}

// Load a 16x32 f16 fragment (A or B layout; lane&15 selects the M/N row,
// K is striped: halves 0..7 -> K = g*8..g*8+7, halves 8..15 -> K = 16+g*8..).
// base points at element [row0][k0] of a row-major matrix with leading dim ld.
static __device__ __forceinline__ v16h frag_load(const _Float16* base, int ld) {
  const int l = threadIdx.x & 31;
  const _Float16* p = base + (size_t)(l & 15) * ld + ((l >> 4) * 8);
  v16h r;
  ((v8h*)&r)[0] = *(const v8h*)(p);
  ((v8h*)&r)[1] = *(const v8h*)(p + 16);
  return r;
}

// ---- 1. RMSNorm -> f16 -----------------------------------------------------

__global__ void rmsnorm_kernel(const float* __restrict__ x,
                               const float* __restrict__ w,
                               _Float16* __restrict__ h16) {
  __shared__ float red[256];
  const int row = blockIdx.x;
  const float* xr = x + (size_t)row * D_MODEL;
  float ss = 0.f;
  for (int i = threadIdx.x; i < D_MODEL; i += 256) { float v = xr[i]; ss += v * v; }
  red[threadIdx.x] = ss;
  __syncthreads();
  for (int s = 128; s > 0; s >>= 1) {
    if (threadIdx.x < s) red[threadIdx.x] += red[threadIdx.x + s];
    __syncthreads();
  }
  const float scale = rsqrtf(red[0] / (float)D_MODEL + 1e-6f);
  for (int i = threadIdx.x; i < D_MODEL; i += 256)
    h16[(size_t)row * D_MODEL + i] = (_Float16)(xr[i] * scale * w[i]);
}

// ---- 2. Weight cast + transpose: Wt[n][k] = (f16) W[k][n] ------------------

__global__ void wcastT_kernel(const float* __restrict__ W,
                              _Float16* __restrict__ Wt, int K, int N) {
  int idx = blockIdx.x * 256 + threadIdx.x;
  if (idx >= K * N) return;
  int k = idx / N, n = idx - k * N;
  Wt[(size_t)n * K + k] = (_Float16)W[idx];
}

// ---- 3. WMMA GEMM: C[M][N] = A[M][K] * Bt[N][K]^T (+ resid) ----------------
// block = 256 thr = 8 waves arranged 4(M) x 2(N); wave tile 32x32; block 128x64.

__global__ void __launch_bounds__(256)
gemm_kernel(const _Float16* __restrict__ A, const _Float16* __restrict__ Bt,
            float* __restrict__ C, const float* __restrict__ resid,
            int M, int N, int K) {
  const int wid = threadIdx.x >> 5;
  const int l = threadIdx.x & 31;
  const int g = l >> 4;
  const int m0 = blockIdx.y * 128 + (wid & 3) * 32;
  const int n0 = blockIdx.x * 64 + (wid >> 2) * 32;
  v8f acc00 = {}, acc01 = {}, acc10 = {}, acc11 = {};
  for (int kk = 0; kk < K; kk += 32) {
    v16h a0 = frag_load(A + (size_t)m0 * K + kk, K);
    v16h a1 = frag_load(A + (size_t)(m0 + 16) * K + kk, K);
    v16h b0 = frag_load(Bt + (size_t)n0 * K + kk, K);
    v16h b1 = frag_load(Bt + (size_t)(n0 + 16) * K + kk, K);
    acc00 = wmma16(a0, b0, acc00);
    acc01 = wmma16(a0, b1, acc01);
    acc10 = wmma16(a1, b0, acc10);
    acc11 = wmma16(a1, b1, acc11);
  }
  v8f* accs[4] = { &acc00, &acc01, &acc10, &acc11 };
  #pragma unroll
  for (int mi = 0; mi < 2; mi++)
    #pragma unroll
    for (int ni = 0; ni < 2; ni++) {
      v8f a = *accs[mi * 2 + ni];
      #pragma unroll
      for (int v = 0; v < 8; v++) {
        int row = m0 + mi * 16 + v + 8 * g;
        int col = n0 + ni * 16 + (l & 15);
        float val = a[v];
        if (resid) val += resid[(size_t)row * N + col];
        C[(size_t)row * N + col] = val;
      }
    }
}

// ---- 4a. Taylor feature map: phi(q@Wqf), phi(k@Wkf) ------------------------
// Outputs: q_phi,k_phi [b][h][t][FPAD] f16 (zero padded), kT [b][h][FPAD][t].

__global__ void featmap_kernel(const float* __restrict__ q32,
                               const float* __restrict__ k32,
                               const float* __restrict__ Wqf,
                               const float* __restrict__ Wkf,
                               _Float16* __restrict__ q_phi,
                               _Float16* __restrict__ k_phi,
                               _Float16* __restrict__ kT) {
  __shared__ float qrow[D_MODEL], krow[D_MODEL];
  __shared__ float tq[N_HEADS][FEAT], tk[N_HEADS][FEAT];
  const int row = blockIdx.x;
  const int b = row >> 11, t = row & (TSEQ - 1);
  for (int i = threadIdx.x; i < D_MODEL; i += 256) {
    qrow[i] = q32[(size_t)row * D_MODEL + i];
    krow[i] = k32[(size_t)row * D_MODEL + i];
  }
  __syncthreads();
  {
    const int h = threadIdx.x >> 4, f = threadIdx.x & 15;
    float sq = 0.f, sk = 0.f;
    for (int d = 0; d < HEAD_DIM; d++) {
      float wq = Wqf[d * FEAT + f], wk = Wkf[d * FEAT + f];
      sq += qrow[h * HEAD_DIM + d] * wq;
      sk += krow[h * HEAD_DIM + d] * wk;
    }
    tq[h][f] = sq;
    tk[h][f] = sk;
  }
  __syncthreads();
  for (int o = threadIdx.x; o < N_HEADS * FPAD; o += 256) {
    int hh = o / FPAD, c = o - hh * FPAD;
    float vq, vk;
    if (c == 0) { vq = 1.f; vk = 1.f; }
    else if (c <= FEAT) { vq = tq[hh][c - 1]; vk = tk[hh][c - 1]; }
    else if (c < 153) {
      int rem = c - 17, i = 0;
      while (rem >= FEAT - i) { rem -= FEAT - i; i++; }
      int j = i + rem;
      float sc = (i == j) ? 0.5f : 0.70710678118654752f;
      vq = tq[hh][i] * tq[hh][j] * sc;
      vk = tk[hh][i] * tk[hh][j] * sc;
    } else { vq = 0.f; vk = 0.f; }
    size_t base = ((size_t)(b * N_HEADS + hh) * TSEQ + t) * FPAD + c;
    q_phi[base] = (_Float16)vq;
    k_phi[base] = (_Float16)vk;
    kT[((size_t)(b * N_HEADS + hh) * FPAD + c) * TSEQ + t] = (_Float16)vk;
  }
}

// ---- 4b. Head-major reorg: qh,kh [b][h][t][64] f16; vT [b][h][DAUG][t] -----
// vT row 64 = 1.0 (ones column -> denominator z via WMMA), rows 65..79 = 0.

__global__ void reorg_kernel(const float* __restrict__ q32,
                             const float* __restrict__ k32,
                             const float* __restrict__ v32,
                             _Float16* __restrict__ qh,
                             _Float16* __restrict__ kh,
                             _Float16* __restrict__ vT) {
  size_t idx = (size_t)blockIdx.x * 256 + threadIdx.x;
  if (idx >= (size_t)BATCH * N_HEADS * DAUG * TSEQ) return;
  int t = (int)(idx & (TSEQ - 1));
  size_t r = idx >> 11;
  int dd = (int)(r % DAUG); r /= DAUG;
  int h = (int)(r % N_HEADS);
  int b = (int)(r / N_HEADS);
  float vv;
  if (dd < HEAD_DIM) {
    size_t src = ((size_t)(b * TSEQ + t)) * D_MODEL + h * HEAD_DIM + dd;
    vv = v32[src];
    size_t ho = ((size_t)(b * N_HEADS + h) * TSEQ + t) * HEAD_DIM + dd;
    qh[ho] = (_Float16)q32[src];
    kh[ho] = (_Float16)k32[src];
  } else {
    vv = (dd == HEAD_DIM) ? 1.f : 0.f;
  }
  vT[((size_t)(b * N_HEADS + h) * DAUG + dd) * TSEQ + t] = (_Float16)vv;
}

// ---- 5. Chunked causal linear attention (one block per (b,h)) --------------
// Per chunk: S = q_phi @ k_phi^T (masked), y = S@V_aug + q_phi@KVT,
// KVT += V_aug^T @ k_phi (accumulated in WMMA registers across chunks).

__global__ void __launch_bounds__(256)
linattn_kernel(const _Float16* __restrict__ q_phi,
               const _Float16* __restrict__ k_phi,
               const _Float16* __restrict__ kT,
               const _Float16* __restrict__ vT,
               _Float16* __restrict__ cc) {
  __shared__ __attribute__((aligned(16))) _Float16 S_lds[CHUNK][CHUNK];   // 32 KB
  __shared__ __attribute__((aligned(16))) _Float16 KVT_lds[DAUG][FPAD];   // 25.6 KB
  __shared__ float z_lds[8][16];

  const int bh = blockIdx.x;                 // 0..31
  const int b = bh >> 4, h = bh & 15;
  const _Float16* qp  = q_phi + (size_t)bh * TSEQ * FPAD;
  const _Float16* kp  = k_phi + (size_t)bh * TSEQ * FPAD;
  const _Float16* ktp = kT    + (size_t)bh * FPAD * TSEQ;
  const _Float16* vtp = vT    + (size_t)bh * DAUG * TSEQ;
  const int w = threadIdx.x >> 5, l = threadIdx.x & 31, g = l >> 4;

  for (int i = threadIdx.x; i < DAUG * FPAD; i += 256)
    (&KVT_lds[0][0])[i] = (_Float16)0.f;
  __syncthreads();

  v8f kvacc[7];
  #pragma unroll
  for (int i = 0; i < 7; i++) kvacc[i] = (v8f){};

  for (int ch = 0; ch < TSEQ / CHUNK; ch++) {
    const int c0 = ch * CHUNK + w * 16;      // global row base of this wave's queries

    v16h aq[FPAD / 32];
    #pragma unroll
    for (int kk = 0; kk < FPAD / 32; kk++)
      aq[kk] = frag_load(qp + (size_t)c0 * FPAD + kk * 32, FPAD);

    // Stage 1: intra-chunk scores, causal mask, store f16 to LDS (A layout)
    for (int n = 0; n < CHUNK / 16; n++) {
      v8f acc = {};
      const int s0 = ch * CHUNK + n * 16;
      #pragma unroll
      for (int kk = 0; kk < FPAD / 32; kk++)
        acc = wmma16(aq[kk], frag_load(kp + (size_t)s0 * FPAD + kk * 32, FPAD), acc);
      #pragma unroll
      for (int v = 0; v < 8; v++) {
        int c = w * 16 + v + 8 * g;          // chunk-local query row
        int s = n * 16 + (l & 15);           // chunk-local key row
        S_lds[c][s] = (_Float16)((s <= c) ? acc[v] : 0.f);
      }
    }
    __syncthreads();

    // Stage 2: y_aug(16 x DAUG) = S @ V_aug + q_phi @ KVT (old state)
    v8f yacc[DAUG / 16];
    #pragma unroll
    for (int n2 = 0; n2 < DAUG / 16; n2++) {
      v8f acc = {};
      #pragma unroll
      for (int kk = 0; kk < CHUNK / 32; kk++) {
        v16h afr = frag_load(&S_lds[w * 16][kk * 32], CHUNK);
        v16h bfr = frag_load(vtp + (size_t)(n2 * 16) * TSEQ + ch * CHUNK + kk * 32, TSEQ);
        acc = wmma16(afr, bfr, acc);
      }
      #pragma unroll
      for (int kk = 0; kk < FPAD / 32; kk++)
        acc = wmma16(aq[kk], frag_load(&KVT_lds[n2 * 16][kk * 32], FPAD), acc);
      yacc[n2] = acc;
    }
    if ((l & 15) == 0) {                      // denominator z = column 64
      #pragma unroll
      for (int v = 0; v < 8; v++) z_lds[w][v + 8 * g] = yacc[4][v];
    }
    __syncthreads();                          // all reads of old KVT_lds done

    // Stage 3: KVT(DAUG x FPAD) += V_aug^T @ k_phi; mirror new state to LDS
    #pragma unroll
    for (int i = 0; i < 7; i++) {
      int tt = w + i * 8;
      if (tt < 50) {                          // wave-uniform branch
        int mi = tt / 10, ni = tt - mi * 10;
        v8f acc = kvacc[i];
        #pragma unroll
        for (int kk = 0; kk < CHUNK / 32; kk++) {
          v16h afr = frag_load(vtp + (size_t)(mi * 16) * TSEQ + ch * CHUNK + kk * 32, TSEQ);
          v16h bfr = frag_load(ktp + (size_t)(ni * 16) * TSEQ + ch * CHUNK + kk * 32, TSEQ);
          acc = wmma16(afr, bfr, acc);
        }
        kvacc[i] = acc;
        #pragma unroll
        for (int v = 0; v < 8; v++)
          KVT_lds[mi * 16 + v + 8 * g][ni * 16 + (l & 15)] = (_Float16)acc[v];
      }
    }

    // Stage 4: normalize and emit lin half of concat buffer (f16)
    #pragma unroll
    for (int n2 = 0; n2 < 4; n2++) {
      #pragma unroll
      for (int v = 0; v < 8; v++) {
        int rloc = v + 8 * g;
        float z = z_lds[w][rloc] + 1e-6f;
        int t = ch * CHUNK + w * 16 + rloc;
        int col = h * HEAD_DIM + n2 * 16 + (l & 15);
        cc[((size_t)(b * TSEQ) + t) * (2 * D_MODEL) + col] = (_Float16)(yacc[n2][v] / z);
      }
    }
    __syncthreads();                          // new KVT_lds visible next chunk
  }
}

// ---- 6. Sliding-window attention (block per (b,h,q-tile of 128)) -----------

__global__ void __launch_bounds__(256)
winattn_kernel(const _Float16* __restrict__ qh, const _Float16* __restrict__ kh,
               const _Float16* __restrict__ vT, _Float16* __restrict__ cc) {
  __shared__ float sc[8][16][80];                                       // 40 KB
  __shared__ __attribute__((aligned(16))) _Float16 P[8][16][96];        // 24 KB
  const int tile = blockIdx.x & 15, bh = blockIdx.x >> 4;
  const int b = bh >> 4, h = bh & 15;
  const _Float16* qb = qh + (size_t)bh * TSEQ * HEAD_DIM;
  const _Float16* kb = kh + (size_t)bh * TSEQ * HEAD_DIM;
  const _Float16* vb = vT + (size_t)bh * DAUG * TSEQ;
  const int w = threadIdx.x >> 5, l = threadIdx.x & 31, g = l >> 4;
  const int q0 = tile * 128 + w * 16;
  const int s0 = q0 - WINDOW;                 // 16-aligned, may be negative
  const float scale = 0.125f;                 // 1/sqrt(64)

  // scores over 5 key tiles (80 keys)
  for (int n = 0; n < 5; n++) {
    v8f acc = {};
    #pragma unroll
    for (int kk = 0; kk < HEAD_DIM; kk += 32) {
      v16h a = frag_load(qb + (size_t)q0 * HEAD_DIM + kk, HEAD_DIM);
      int r = s0 + n * 16 + (l & 15);
      if (r < 0) r = 0;                       // masked later
      const _Float16* p = kb + (size_t)r * HEAD_DIM + kk + g * 8;
      v16h bfr;
      ((v8h*)&bfr)[0] = *(const v8h*)(p);
      ((v8h*)&bfr)[1] = *(const v8h*)(p + 16);
      acc = wmma16(a, bfr, acc);
    }
    #pragma unroll
    for (int v = 0; v < 8; v++) {
      int c = q0 + v + 8 * g;
      int s = s0 + n * 16 + (l & 15);
      bool ok = (s >= 0) && (s <= c) && (s >= c - WINDOW);
      sc[w][v + 8 * g][n * 16 + (l & 15)] = ok ? acc[v] * scale : -1e30f;
    }
  }
  __syncthreads();

  // row softmax (lanes 0..15, one row each)
  if (l < 16) {
    float m = -1e30f;
    for (int j = 0; j < 80; j++) m = fmaxf(m, sc[w][l][j]);
    float sum = 0.f;
    for (int j = 0; j < 80; j++) sum += __expf(sc[w][l][j] - m);
    float inv = 1.f / sum;
    for (int j = 0; j < 80; j++) P[w][l][j] = (_Float16)(__expf(sc[w][l][j] - m) * inv);
    for (int j = 80; j < 96; j++) P[w][l][j] = (_Float16)0.f;
  }
  __syncthreads();

  // out = P(16x96) @ V(96x64)  (zero-padded P cols kill over-read tail)
  #pragma unroll
  for (int n2 = 0; n2 < 4; n2++) {
    v8f acc = {};
    #pragma unroll
    for (int kk = 0; kk < 3; kk++) {
      v16h a = frag_load(&P[w][0][kk * 32], 96);
      int d = n2 * 16 + (l & 15);
      const _Float16* rowp = vb + (size_t)d * TSEQ;
      int ta = s0 + kk * 32 + g * 8;
      int tb = ta + 16;
      if (ta < 0) ta = 0;
      if (tb < 0) tb = 0;
      v16h bfr;
      ((v8h*)&bfr)[0] = *(const v8h*)(rowp + ta);
      ((v8h*)&bfr)[1] = *(const v8h*)(rowp + tb);
      acc = wmma16(a, bfr, acc);
    }
    #pragma unroll
    for (int v = 0; v < 8; v++) {
      int t = q0 + v + 8 * g;
      int col = D_MODEL + h * HEAD_DIM + n2 * 16 + (l & 15);
      cc[((size_t)(b * TSEQ) + t) * (2 * D_MODEL) + col] = (_Float16)acc[v];
    }
  }
}

// ---- host launch -----------------------------------------------------------

extern "C" void kernel_launch(void* const* d_in, const int* in_sizes, int n_in,
                              void* d_out, int out_size, void* d_ws, size_t ws_size,
                              hipStream_t stream) {
  const float* x      = (const float*)d_in[0];
  const float* norm_w = (const float*)d_in[1];
  const float* Wq     = (const float*)d_in[2];
  const float* Wk     = (const float*)d_in[3];
  const float* Wv     = (const float*)d_in[4];
  const float* Wqf    = (const float*)d_in[5];
  const float* Wkf    = (const float*)d_in[6];
  const float* Wout   = (const float*)d_in[7];
  float* out = (float*)d_out;

  char* p = (char*)d_ws;
  auto alloc = [&](size_t bytes) -> char* {
    char* r = p;
    p += (bytes + 255) & ~(size_t)255;
    return r;
  };
  _Float16* h16   = (_Float16*)alloc((size_t)NROWS * D_MODEL * 2);
  _Float16* wqT   = (_Float16*)alloc((size_t)D_MODEL * D_MODEL * 2);
  _Float16* wkT   = (_Float16*)alloc((size_t)D_MODEL * D_MODEL * 2);
  _Float16* wvT   = (_Float16*)alloc((size_t)D_MODEL * D_MODEL * 2);
  _Float16* woT   = (_Float16*)alloc((size_t)D_MODEL * 2 * D_MODEL * 2);
  float*    q32   = (float*)alloc((size_t)NROWS * D_MODEL * 4);
  float*    k32   = (float*)alloc((size_t)NROWS * D_MODEL * 4);
  float*    v32   = (float*)alloc((size_t)NROWS * D_MODEL * 4);
  _Float16* q_phi = (_Float16*)alloc((size_t)BATCH * N_HEADS * TSEQ * FPAD * 2);
  _Float16* k_phi = (_Float16*)alloc((size_t)BATCH * N_HEADS * TSEQ * FPAD * 2);
  _Float16* kT    = (_Float16*)alloc((size_t)BATCH * N_HEADS * FPAD * TSEQ * 2);
  _Float16* qhh   = (_Float16*)alloc((size_t)BATCH * N_HEADS * TSEQ * HEAD_DIM * 2);
  _Float16* khh   = (_Float16*)alloc((size_t)BATCH * N_HEADS * TSEQ * HEAD_DIM * 2);
  _Float16* vT    = (_Float16*)alloc((size_t)BATCH * N_HEADS * DAUG * TSEQ * 2);
  _Float16* cc    = (_Float16*)alloc((size_t)NROWS * 2 * D_MODEL * 2);
  (void)ws_size; (void)in_sizes; (void)n_in; (void)out_size;

  // 1. RMSNorm
  rmsnorm_kernel<<<dim3(NROWS), dim3(256), 0, stream>>>(x, norm_w, h16);

  // 2. Weight casts/transposes
  {
    int n1 = D_MODEL * D_MODEL;
    wcastT_kernel<<<dim3((n1 + 255) / 256), dim3(256), 0, stream>>>(Wq, wqT, D_MODEL, D_MODEL);
    wcastT_kernel<<<dim3((n1 + 255) / 256), dim3(256), 0, stream>>>(Wk, wkT, D_MODEL, D_MODEL);
    wcastT_kernel<<<dim3((n1 + 255) / 256), dim3(256), 0, stream>>>(Wv, wvT, D_MODEL, D_MODEL);
    int n2 = 2 * D_MODEL * D_MODEL;
    wcastT_kernel<<<dim3((n2 + 255) / 256), dim3(256), 0, stream>>>(Wout, woT, 2 * D_MODEL, D_MODEL);
  }

  // 3. Q/K/V projections (WMMA GEMM, M=4096 N=1024 K=1024)
  dim3 ggrid(D_MODEL / 64, NROWS / 128);
  gemm_kernel<<<ggrid, dim3(256), 0, stream>>>(h16, wqT, q32, nullptr, NROWS, D_MODEL, D_MODEL);
  gemm_kernel<<<ggrid, dim3(256), 0, stream>>>(h16, wkT, k32, nullptr, NROWS, D_MODEL, D_MODEL);
  gemm_kernel<<<ggrid, dim3(256), 0, stream>>>(h16, wvT, v32, nullptr, NROWS, D_MODEL, D_MODEL);

  // 4. Feature map + head-major reorg
  featmap_kernel<<<dim3(NROWS), dim3(256), 0, stream>>>(q32, k32, Wqf, Wkf, q_phi, k_phi, kT);
  {
    size_t tot = (size_t)BATCH * N_HEADS * DAUG * TSEQ;
    reorg_kernel<<<dim3((unsigned)((tot + 255) / 256)), dim3(256), 0, stream>>>(
        q32, k32, v32, qhh, khh, vT);
  }

  // 5. Linear attention (writes cc[:, 0:1024])
  linattn_kernel<<<dim3(BATCH * N_HEADS), dim3(256), 0, stream>>>(q_phi, k_phi, kT, vT, cc);

  // 6. Sliding-window attention (writes cc[:, 1024:2048])
  winattn_kernel<<<dim3(BATCH * N_HEADS * (TSEQ / 128)), dim3(256), 0, stream>>>(qhh, khh, vT, cc);

  // 7. Output projection + residual: out = x + cc @ Wout
  gemm_kernel<<<dim3(D_MODEL / 64, NROWS / 128), dim3(256), 0, stream>>>(
      cc, woT, out, x, NROWS, D_MODEL, 2 * D_MODEL);
}
